// CVAE_13486197309950
// MI455X (gfx1250) — compile-verified, matrix operand order checked by reading, not run on hardware
//
#include <hip/hip_runtime.h>
#include <hip/hip_bf16.h>

typedef _Float16 h8  __attribute__((ext_vector_type(8)));
typedef _Float16 h16 __attribute__((ext_vector_type(16)));
typedef float    f4  __attribute__((ext_vector_type(4)));
typedef float    v8f __attribute__((ext_vector_type(8)));

#define HIDDEN   128
#define GATES    512
#define DEC_IN   39     // 2 + 32 + 4 + 1
#define KBLK     5      // K padded to 160: 128 hidden + 2 x + 30 zero
#define NT       32     // 512 gates / 16
#define WAVES    4
#define ROWS_WG  64     // 4 waves * 16 rows
#define STRIDE_H 168    // staging row stride in f16 (160 used; 336B breaks bank conflicts)

// ---- activations: use CDNA5 hardware V_TANH_F32 when clang exposes it ----
#if __has_builtin(__builtin_amdgcn_tanhf)
__device__ __forceinline__ float ftanh(float x) {
  return __builtin_amdgcn_tanhf(x);
}
__device__ __forceinline__ float fsig(float x) {
  // sigma(x) = 0.5*tanh(x/2) + 0.5  -> mul + v_tanh_f32 + fma
  return __builtin_fmaf(0.5f, __builtin_amdgcn_tanhf(0.5f * x), 0.5f);
}
#else
__device__ __forceinline__ float ftanh(float x) {
  float e = __expf(-2.0f * x);
  return (1.0f - e) * __builtin_amdgcn_rcpf(1.0f + e);
}
__device__ __forceinline__ float fsig(float x) {
  return __builtin_amdgcn_rcpf(1.0f + __expf(-x));
}
#endif

__device__ __forceinline__ v8f wmma16(h16 a, h16 b, v8f c) {
  return __builtin_amdgcn_wmma_f32_16x16x32_f16(false, a, false, b, (short)0, c,
                                                false, false);
}

// -------------------------------------------------------------------------
// Prep: static_proj[b,n] = static_in[b,:] @ W_ih[:,2:].T + b_ih + b_hh,
// written into d_ws in WMMA D-fragment layout:
//   region (T,jt) of 256 f32: [load(2)][lane(32)][4 f32]
//   where b = T*16 + (lane>=16)*8 + load*4 + r, n = jt*16 + (lane&15)
// -------------------------------------------------------------------------
__global__ void cvae_prep_static(const float* __restrict__ z,
                                 const float* __restrict__ cond,
                                 const float* __restrict__ W_ih,
                                 const float* __restrict__ b_ih,
                                 const float* __restrict__ b_hh,
                                 float* __restrict__ stat) {
  int tid = blockIdx.x * 256 + threadIdx.x;      // 512*32*32*8 = 4194304 exact
  int rp   = tid & 3;
  int load = (tid >> 2) & 1;
  int ln   = (tid >> 3) & 31;
  int jt   = (tid >> 8) & 31;
  int T    = tid >> 13;
  int b = T * 16 + (ln >> 4) * 8 + load * 4 + rp;
  int n = jt * 16 + (ln & 15);
  const float* wr = W_ih + (size_t)n * DEC_IN;
  float acc = b_ih[n] + b_hh[n] + wr[38] * 0.05f;      // dt column
  #pragma unroll 4
  for (int j = 0; j < 32; ++j) acc += wr[2 + j] * z[(size_t)b * 32 + j];
  #pragma unroll
  for (int j = 0; j < 4; ++j)  acc += wr[34 + j] * cond[(size_t)b * 4 + j];
  stat[(size_t)(T * 32 + jt) * 256 + load * 128 + ln * 4 + rp] = acc;
}

// -------------------------------------------------------------------------
// Main rollout kernel: 128 blocks x 128 threads; each wave owns 16 rows.
// -------------------------------------------------------------------------
__global__ void __launch_bounds__(128)
cvae_lstm_rollout(const float* __restrict__ start_point,
                  const float* __restrict__ W_ih,
                  const float* __restrict__ W_hh,
                  const float* __restrict__ W_out,
                  const float* __restrict__ b_out,
                  const float* __restrict__ stat,
                  float* __restrict__ out,
                  const int* __restrict__ seqp) {
  // B-fragments: 32 N-tiles * 5 K-blocks, each = 2 chunks of 256 f16
  __shared__ __attribute__((aligned(16))) _Float16 ldsB[NT * KBLK * 512];  // 160 KB
  __shared__ __attribute__((aligned(16))) _Float16 ldsWo[4 * 512];         //   4 KB
  __shared__ __attribute__((aligned(16))) _Float16 stage[WAVES * 16 * STRIDE_H]; // 21 KB

  const int tid = threadIdx.x;

  // ---- fill W_hh (+ folded W_ih_x) B-fragments -------------------------
  for (int e = tid; e < NT * KBLK * 512; e += 128) {
    int frag  = e >> 9;            // (jt*5 + kb)
    int within = e & 511;
    int chunk = within >> 8;       // which b128 chunk
    int ln    = (within >> 3) & 31;
    int ii    = within & 7;
    int jt = frag / KBLK, kb = frag % KBLK;
    int n  = jt * 16 + (ln & 15);
    int k  = kb * 32 + ii + chunk * 16 + (ln >> 4) * 8;
    float v;
    if (kb < 4) {
      v = W_hh[(size_t)n * HIDDEN + k];
    } else {
      int kk = k - 128;
      v = (kk < 2) ? W_ih[(size_t)n * DEC_IN + kk] : 0.0f;
    }
    ldsB[e] = (_Float16)v;
  }
  // ---- fill W_out B-fragments (only cols 0,1 valid) --------------------
  for (int e = tid; e < 4 * 512; e += 128) {
    int kb = e >> 9;
    int within = e & 511;
    int chunk = within >> 8;
    int ln = (within >> 3) & 31;
    int ii = within & 7;
    int np = ln & 15;
    int k  = kb * 32 + ii + chunk * 16 + (ln >> 4) * 8;
    ldsWo[e] = (_Float16)((np < 2) ? W_out[(size_t)np * HIDDEN + k] : 0.0f);
  }
  // ---- staging init: h=0, x=start_point, pad=0 -------------------------
  for (int e = tid; e < WAVES * 16 * STRIDE_H; e += 128) {
    int row = e / STRIDE_H, col = e % STRIDE_H;
    float v = 0.0f;
    if (col == 128 || col == 129) {
      int b = blockIdx.x * ROWS_WG + row;
      v = start_point[(size_t)b * 2 + (col - 128)];
    }
    stage[e] = (_Float16)v;
  }
  __syncthreads();

  const int w    = tid >> 5;
  const int ln   = tid & 31;
  const int half = ln >> 4;
  const int np   = ln & 15;
  const int wgrow = blockIdx.x * ROWS_WG + w * 16;
  const int T     = blockIdx.x * WAVES + w;
  const int srow  = w * 16 * STRIDE_H;
  const int seq   = *seqp;
  const float bo  = (np < 2) ? b_out[np] : 0.0f;

  auto loadA = [&](int kb) -> h16 {
    int idx = srow + np * STRIDE_H + kb * 32 + half * 8;
    h8 lo = *(const h8*)&stage[idx];
    h8 hi = *(const h8*)&stage[idx + 16];
    return __builtin_shufflevector(lo, hi, 0,1,2,3,4,5,6,7,8,9,10,11,12,13,14,15);
  };
  auto loadB = [&](int frag) -> h16 {
    int idx = frag * 512 + ln * 8;
    h8 lo = *(const h8*)&ldsB[idx];
    h8 hi = *(const h8*)&ldsB[idx + 256];
    return __builtin_shufflevector(lo, hi, 0,1,2,3,4,5,6,7,8,9,10,11,12,13,14,15);
  };
  auto loadWo = [&](int kb) -> h16 {
    int idx = kb * 512 + ln * 8;
    h8 lo = *(const h8*)&ldsWo[idx];
    h8 hi = *(const h8*)&ldsWo[idx + 256];
    return __builtin_shufflevector(lo, hi, 0,1,2,3,4,5,6,7,8,9,10,11,12,13,14,15);
  };
  auto loadC = [&](int jt) -> v8f {
    size_t base = (size_t)(T * 32 + jt) * 256 + ln * 4;
    f4 lo = *(const f4*)(stat + base);
    f4 hi = *(const f4*)(stat + base + 128);
    return __builtin_shufflevector(lo, hi, 0,1,2,3,4,5,6,7);
  };

  float cst[64];
  #pragma unroll
  for (int i = 0; i < 64; ++i) cst[i] = 0.0f;

  for (int t = 0; t < seq; ++t) {
    // A fragments: [h_{t-1}(128) | x_t(2) | zeros(30)]
    h16 A[KBLK];
    #pragma unroll
    for (int kb = 0; kb < KBLK; ++kb) A[kb] = loadA(kb);

    #pragma unroll
    for (int j = 0; j < 8; ++j) {
      v8f gi = loadC(j);
      v8f gf = loadC(j + 8);
      v8f gg = loadC(j + 16);
      v8f go = loadC(j + 24);
      #pragma unroll
      for (int kb = 0; kb < KBLK; ++kb) {
        gi = wmma16(A[kb], loadB((j)      * KBLK + kb), gi);
        gf = wmma16(A[kb], loadB((j + 8)  * KBLK + kb), gf);
        gg = wmma16(A[kb], loadB((j + 16) * KBLK + kb), gg);
        go = wmma16(A[kb], loadB((j + 24) * KBLK + kb), go);
      }
      #pragma unroll
      for (int r = 0; r < 8; ++r) {
        float iv = fsig(gi[r]);
        float fv = fsig(gf[r]);
        float gv = ftanh(gg[r]);
        float ov = fsig(go[r]);
        float cv = __builtin_fmaf(fv, cst[j * 8 + r], iv * gv);
        cst[j * 8 + r] = cv;
        float hv = ov * ftanh(cv);
        int m = half * 8 + r;
        stage[srow + m * STRIDE_H + j * 16 + np] = (_Float16)hv;
      }
    }

    // x_next = h_t @ W_out.T + b_out  (reload fresh h fragments)
    h16 Ah[4];
    #pragma unroll
    for (int kb = 0; kb < 4; ++kb) Ah[kb] = loadA(kb);
    v8f xa;
    #pragma unroll
    for (int r = 0; r < 8; ++r) xa[r] = bo;
    #pragma unroll
    for (int kb = 0; kb < 4; ++kb) xa = wmma16(Ah[kb], loadWo(kb), xa);

    if (np < 2) {
      #pragma unroll
      for (int r = 0; r < 8; ++r) {
        int m = half * 8 + r;
        int b = wgrow + m;
        out[((size_t)b * seq + t) * 2 + np] = xa[r];
        stage[srow + m * STRIDE_H + 128 + np] = (_Float16)xa[r];
      }
    }
  }
}

extern "C" void kernel_launch(void* const* d_in, const int* in_sizes, int n_in,
                              void* d_out, int out_size, void* d_ws, size_t ws_size,
                              hipStream_t stream) {
  const float* z        = (const float*)d_in[0];
  const float* cond     = (const float*)d_in[1];
  const float* sp       = (const float*)d_in[2];
  const float* W_ih     = (const float*)d_in[3];
  const float* W_hh     = (const float*)d_in[4];
  const float* b_ih     = (const float*)d_in[5];
  const float* b_hh     = (const float*)d_in[6];
  const float* W_out    = (const float*)d_in[7];
  const float* b_out    = (const float*)d_in[8];
  const int*   seqp     = (const int*)d_in[9];
  float* stat = (float*)d_ws;          // 8192*512 f32 = 16.8 MB
  float* out  = (float*)d_out;

  cvae_prep_static<<<16384, 256, 0, stream>>>(z, cond, W_ih, b_ih, b_hh, stat);
  cvae_lstm_rollout<<<128, 128, 0, stream>>>(sp, W_ih, W_hh, W_out, b_out,
                                             stat, out, seqp);
}
